// HierarchicalGRUBatterHistoryEncoder_31997506355483
// MI455X (gfx1250) — compile-verified, matrix operand order checked by reading, not run on hardware
//
#include <hip/hip_runtime.h>
#include <stdint.h>

// ---------------------------------------------------------------------------
// Hierarchical GRU batter-history encoder for gfx1250 (MI455X, wave32, WMMA).
//   B=256, N=64, P=32, H=256, D_IN=29 (pad 32), D_AB=264 (pad 288), gates 3H=768
// bf16 WMMA (v_wmma_f32_16x16x32_bf16), f32 accumulation, f32 register state.
// Tiling: 32 rows/block, 16 waves as 2(M) x 8(N); wave tile = 16 rows x 32 cols
// per gate set => 8 v8f accumulators (64 VGPRs) — no spills.
// LICM on weight-fragment loads is defeated with an opaque SGPR zero folded
// into the *offset* (not the pointer), so loads stay per-step but keep global
// address-space inference => global_load_b128 (LOADcnt only, no flat/DScnt).
// ---------------------------------------------------------------------------

#define BDIM 256
#define NDIM 64
#define PDIM 32
#define ROWS 32            // rows (sequences) per block

typedef __attribute__((ext_vector_type(16))) __bf16 v16bf;
typedef __attribute__((ext_vector_type(8)))  float  v8f;

union Frag {
    v16bf v;
    uint4 q[2];
};

__device__ __forceinline__ uint16_t f2bf(float f) {
    union { float f; uint32_t u; } c; c.f = f;
    uint32_t u = c.u;
    return (uint16_t)((u + 0x7FFFu + ((u >> 16) & 1u)) >> 16);
}

__device__ __forceinline__ float sigmoidf_(float x) {
    return 1.0f / (1.0f + __expf(-x));
}

// Opaque zero in an SGPR: added into addressing each time step so fragment
// loads are loop-variant (no LICM hoist -> no spill), while the base pointer
// stays kernarg-derived (global address space -> global_load, not flat_load).
__device__ __forceinline__ int step_token() {
    int z = 0;
    asm volatile("" : "+s"(z));
    return z;
}

// A-fragment (16x32 bf16, M x K), row-major source with arbitrary row stride.
// ISA layout: lanes 0-15 -> rows, khalf = lane>>4; elems 0..7 -> k = khalf*8+e,
// elems 8..15 -> k = khalf*8+16+(e-8)  => two contiguous 16B loads per lane.
__device__ __forceinline__ Frag load_a_frag(const uint16_t* base, size_t rowStride,
                                            int rowBase, int kBase) {
    const int lane  = threadIdx.x & 31;
    const int row   = rowBase + (lane & 15);
    const int khalf = lane >> 4;
    const uint16_t* p = base + (size_t)row * rowStride + kBase + khalf * 8;
    Frag f;
    f.q[0] = *(const uint4*)(p);
    f.q[1] = *(const uint4*)(p + 16);
    return f;
}

// B-fragment from pre-swizzled weight buffer: 512 bf16 per 32x16 tile,
// stored [tile][lane][16] so each lane reads 32 contiguous bytes.
__device__ __forceinline__ Frag load_b_frag(const uint16_t* frags, int tileIdx) {
    const int lane = threadIdx.x & 31;
    const uint16_t* p = frags + (size_t)tileIdx * 512 + lane * 16;
    Frag f;
    f.q[0] = *(const uint4*)(p);
    f.q[1] = *(const uint4*)(p + 8);
    return f;
}

__device__ __forceinline__ v8f wmma_acc(v8f c, const Frag& a, const Frag& b) {
    return __builtin_amdgcn_wmma_f32_16x16x32_bf16(
        /*neg_a=*/false, a.v, /*neg_b=*/false, b.v,
        /*c_mod=*/(short)0, c, /*reuse_a=*/false, /*reuse_b=*/false);
}

// ---------------------------------------------------------------------------
// Weight swizzle: W [768 x Kreal] f32 row-major -> bf16 B-fragments
// dst layout: [ntile(48)][ktile(KT)][lane(32)][elem(16)], k padded with zeros.
// ---------------------------------------------------------------------------
__global__ void swizzle_weight_kernel(const float* __restrict__ W, int Kreal,
                                      int KT, uint16_t* __restrict__ dst, int total) {
    int idx = blockIdx.x * blockDim.x + threadIdx.x;
    if (idx >= total) return;
    int e     = idx & 15;
    int lane  = (idx >> 4) & 31;
    int tile  = idx >> 9;
    int kt    = tile % KT;
    int ntile = tile / KT;
    int n = ntile * 16 + (lane & 15);
    int k = kt * 32 + (lane >> 4) * 8 + (e < 8 ? e : e + 8);
    float v = (k < Kreal) ? W[(size_t)n * Kreal + k] : 0.0f;
    dst[idx] = f2bf(v);
}

// ---------------------------------------------------------------------------
// Inner GRU: 16384 sequences, 32 steps. One block = 32 sequences, 16 waves
// arranged 2(M) x 8(N): wave tile = 16 rows x 32 hidden units (x3 gates).
// ---------------------------------------------------------------------------
__global__ __launch_bounds__(512, 1) void inner_gru_kernel(
    const int* __restrict__ pt_idx, const float* __restrict__ cont,
    const float* __restrict__ swing, const int* __restrict__ sr_idx,
    const float* __restrict__ pmask,
    const float* __restrict__ pt_embed, const float* __restrict__ sr_embed,
    const uint16_t* __restrict__ wih, const uint16_t* __restrict__ whh,
    const float* __restrict__ b_ih, const float* __restrict__ b_hh,
    uint16_t* __restrict__ inner_out)
{
    __shared__ __align__(16) uint16_t h_bf[ROWS * 264];  // bf16 state, padded rows
    __shared__ __align__(16) uint16_t x_bf[ROWS * 40];   // step features (29 -> 32)
    __shared__ float mask_s[ROWS];

    const int tid  = threadIdx.x;
    const int wave = tid >> 5;
    const int lane = tid & 31;
    const int mi = wave >> 3;      // row group 0..1
    const int ni = wave & 7;       // col group 0..7
    const int rowBase = mi * 16;
    const int seq0 = blockIdx.x * ROWS;

    for (int i = tid; i < ROWS * 264; i += 512) h_bf[i] = 0;
    __syncthreads();

    v8f hreg[2];
#pragma unroll
    for (int jj = 0; jj < 2; ++jj)
#pragma unroll
        for (int e = 0; e < 8; ++e) hreg[jj][e] = 0.0f;

#pragma unroll 1
    for (int t = 0; t < PDIM; ++t) {
        // -------- build x_t features (pt_embed | cont | swing | sr_embed | 0)
        for (int idx = tid; idx < ROWS * 32; idx += 512) {
            int r = idx >> 5, j = idx & 31;
            int s = seq0 + r;
            float v = 0.0f;
            if (j < 16) {
                int p = pt_idx[s * PDIM + t];
                p = p < 0 ? 0 : (p > 20 ? 20 : p);
                v = pt_embed[p * 16 + j];
            } else if (j < 24) {
                v = cont[((size_t)s * PDIM + t) * 8 + (j - 16)];
            } else if (j == 24) {
                v = swing[s * PDIM + t];
            } else if (j < 29) {
                int q = sr_idx[s * PDIM + t];
                if (q < 0) q = 8;
                v = sr_embed[q * 4 + (j - 25)];
            }
            x_bf[r * 40 + j] = f2bf(v);
        }
        if (tid < ROWS) mask_s[tid] = pmask[(seq0 + tid) * PDIM + t];
        __syncthreads();

        // per-step opaque offset: weight loads stay inside the loop but keep
        // global address-space inference (global_load_b128, LOADcnt only)
        const int zo = step_token();
        const uint16_t* wih_t = wih + zo;
        const uint16_t* whh_t = whh + zo;

        // -------- accumulators seeded with biases
        v8f accR[2], accZ[2], accNX[2], accNH[2];
#pragma unroll
        for (int jj = 0; jj < 2; ++jj) {
            int n = (ni * 2 + jj) * 16 + (lane & 15);
            float br  = b_ih[n] + b_hh[n];
            float bz  = b_ih[256 + n] + b_hh[256 + n];
            float bnx = b_ih[512 + n];
            float bnh = b_hh[512 + n];
#pragma unroll
            for (int e = 0; e < 8; ++e) {
                accR[jj][e] = br;  accZ[jj][e] = bz;
                accNX[jj][e] = bnx; accNH[jj][e] = bnh;
            }
        }

        // -------- input projection (single K-tile, K=29 padded to 32)
        {
            Frag ax = load_a_frag(x_bf, 40, rowBase, 0);
#pragma unroll
            for (int jj = 0; jj < 2; ++jj) {
                int jt = ni * 2 + jj;
                accR[jj]  = wmma_acc(accR[jj],  ax, load_b_frag(wih_t, (0 * 16 + jt)));
                accZ[jj]  = wmma_acc(accZ[jj],  ax, load_b_frag(wih_t, (1 * 16 + jt)));
                accNX[jj] = wmma_acc(accNX[jj], ax, load_b_frag(wih_t, (2 * 16 + jt)));
            }
        }
        // -------- hidden recurrence (8 K-tiles over H=256)
#pragma unroll
        for (int kt = 0; kt < 8; ++kt) {
            Frag ah = load_a_frag(h_bf, 264, rowBase, kt * 32);
#pragma unroll
            for (int jj = 0; jj < 2; ++jj) {
                int jt = ni * 2 + jj;
                accR[jj]  = wmma_acc(accR[jj],  ah, load_b_frag(whh_t, (0 * 16 + jt) * 8 + kt));
                accZ[jj]  = wmma_acc(accZ[jj],  ah, load_b_frag(whh_t, (1 * 16 + jt) * 8 + kt));
                accNH[jj] = wmma_acc(accNH[jj], ah, load_b_frag(whh_t, (2 * 16 + jt) * 8 + kt));
            }
        }
        __syncthreads();   // all reads of h_bf / x_bf complete

        // -------- gate nonlinearity + masked state update (pure registers)
#pragma unroll
        for (int jj = 0; jj < 2; ++jj) {
            int col = (ni * 2 + jj) * 16 + (lane & 15);
#pragma unroll
            for (int e = 0; e < 8; ++e) {
                int row = rowBase + e + ((lane >> 4) << 3);
                float hprev = hreg[jj][e];
                float rg = sigmoidf_(accR[jj][e]);
                float zg = sigmoidf_(accZ[jj][e]);
                float ng = tanhf(accNX[jj][e] + rg * accNH[jj][e]);
                float hn = (1.0f - zg) * ng + zg * hprev;
                float hv = (mask_s[row] > 0.0f) ? hn : hprev;
                hreg[jj][e] = hv;
                h_bf[row * 264 + col] = f2bf(hv);
            }
        }
        __syncthreads();   // h_bf updated for next step
    }

    // -------- write inner_out [seq][256] bf16 (vectorized 16B copies)
    for (int i = tid; i < ROWS * 32; i += 512) {
        int r = i >> 5, c = i & 31;
        ((uint4*)(inner_out + (size_t)(seq0 + r) * 256))[c] =
            ((const uint4*)(h_bf + r * 264))[c];
    }
}

// ---------------------------------------------------------------------------
// Outer GRU layer (used twice): 256 rows, 64 steps. 8 blocks x 512 threads.
// x sequence (256-dim bf16) streamed from global; optional 32-dim extras tile.
// ---------------------------------------------------------------------------
__global__ __launch_bounds__(512, 1) void outer_gru_kernel(
    const uint16_t* __restrict__ xseq,           // [B*64][256] bf16
    const int* __restrict__ bb_idx, const float* __restrict__ bb_embed,
    const float* __restrict__ lspeed, const float* __restrict__ langle,
    const float* __restrict__ hc_x, const float* __restrict__ hc_y,
    int has_extra,
    const float* __restrict__ amask,
    const uint16_t* __restrict__ wih, int ktx,   // K-tiles in W_ih (9 or 8)
    const uint16_t* __restrict__ whh,
    const float* __restrict__ b_ih, const float* __restrict__ b_hh,
    uint16_t* __restrict__ seq_out,              // nullable
    float* __restrict__ final_out)               // nullable
{
    __shared__ __align__(16) uint16_t h_bf[ROWS * 264];
    __shared__ __align__(16) uint16_t x_bf[ROWS * 40];
    __shared__ float mask_s[ROWS];

    const int tid  = threadIdx.x;
    const int wave = tid >> 5;
    const int lane = tid & 31;
    const int mi = wave >> 3;
    const int ni = wave & 7;
    const int rowBase = mi * 16;
    const int b0 = blockIdx.x * ROWS;

    for (int i = tid; i < ROWS * 264; i += 512) h_bf[i] = 0;
    __syncthreads();

    v8f hreg[2];
#pragma unroll
    for (int jj = 0; jj < 2; ++jj)
#pragma unroll
        for (int e = 0; e < 8; ++e) hreg[jj][e] = 0.0f;

#pragma unroll 1
    for (int t = 0; t < NDIM; ++t) {
        if (has_extra) {
            for (int idx = tid; idx < ROWS * 32; idx += 512) {
                int r = idx >> 5, c = idx & 31;
                int s = (b0 + r) * NDIM + t;
                float v = 0.0f;
                if (c < 4) {
                    int q = bb_idx[s]; if (q < 0) q = 8;
                    v = bb_embed[q * 4 + c];
                } else if (c == 4) v = lspeed[s];
                else if (c == 5) v = langle[s];
                else if (c == 6) v = hc_x[s];
                else if (c == 7) v = hc_y[s];
                x_bf[r * 40 + c] = f2bf(v);
            }
        }
        if (tid < ROWS) mask_s[tid] = amask[(b0 + tid) * NDIM + t];
        __syncthreads();

        const int zo = step_token();
        const uint16_t* wih_t = wih + zo;
        const uint16_t* whh_t = whh + zo;

        v8f accR[2], accZ[2], accNX[2], accNH[2];
#pragma unroll
        for (int jj = 0; jj < 2; ++jj) {
            int n = (ni * 2 + jj) * 16 + (lane & 15);
            float br  = b_ih[n] + b_hh[n];
            float bz  = b_ih[256 + n] + b_hh[256 + n];
            float bnx = b_ih[512 + n];
            float bnh = b_hh[512 + n];
#pragma unroll
            for (int e = 0; e < 8; ++e) {
                accR[jj][e] = br;  accZ[jj][e] = bz;
                accNX[jj][e] = bnx; accNH[jj][e] = bnh;
            }
        }

        // x projection: 8 K-tiles streamed from global sequence input
        const uint16_t* xbase = xseq + zo + ((size_t)b0 * NDIM + t) * 256;
#pragma unroll
        for (int kt = 0; kt < 8; ++kt) {
            Frag ax = load_a_frag(xbase, (size_t)NDIM * 256, rowBase, kt * 32);
#pragma unroll
            for (int jj = 0; jj < 2; ++jj) {
                int jt = ni * 2 + jj;
                accR[jj]  = wmma_acc(accR[jj],  ax, load_b_frag(wih_t, (0 * 16 + jt) * ktx + kt));
                accZ[jj]  = wmma_acc(accZ[jj],  ax, load_b_frag(wih_t, (1 * 16 + jt) * ktx + kt));
                accNX[jj] = wmma_acc(accNX[jj], ax, load_b_frag(wih_t, (2 * 16 + jt) * ktx + kt));
            }
        }
        if (has_extra) {  // extras K-tile (dims 256..287 of D_AB, zero-padded)
            Frag ax = load_a_frag(x_bf, 40, rowBase, 0);
#pragma unroll
            for (int jj = 0; jj < 2; ++jj) {
                int jt = ni * 2 + jj;
                accR[jj]  = wmma_acc(accR[jj],  ax, load_b_frag(wih_t, (0 * 16 + jt) * ktx + 8));
                accZ[jj]  = wmma_acc(accZ[jj],  ax, load_b_frag(wih_t, (1 * 16 + jt) * ktx + 8));
                accNX[jj] = wmma_acc(accNX[jj], ax, load_b_frag(wih_t, (2 * 16 + jt) * ktx + 8));
            }
        }
        // hidden recurrence
#pragma unroll
        for (int kt = 0; kt < 8; ++kt) {
            Frag ah = load_a_frag(h_bf, 264, rowBase, kt * 32);
#pragma unroll
            for (int jj = 0; jj < 2; ++jj) {
                int jt = ni * 2 + jj;
                accR[jj]  = wmma_acc(accR[jj],  ah, load_b_frag(whh_t, (0 * 16 + jt) * 8 + kt));
                accZ[jj]  = wmma_acc(accZ[jj],  ah, load_b_frag(whh_t, (1 * 16 + jt) * 8 + kt));
                accNH[jj] = wmma_acc(accNH[jj], ah, load_b_frag(whh_t, (2 * 16 + jt) * 8 + kt));
            }
        }
        __syncthreads();

#pragma unroll
        for (int jj = 0; jj < 2; ++jj) {
            int col = (ni * 2 + jj) * 16 + (lane & 15);
#pragma unroll
            for (int e = 0; e < 8; ++e) {
                int row = rowBase + e + ((lane >> 4) << 3);
                float hprev = hreg[jj][e];
                float rg = sigmoidf_(accR[jj][e]);
                float zg = sigmoidf_(accZ[jj][e]);
                float ng = tanhf(accNX[jj][e] + rg * accNH[jj][e]);
                float hn = (1.0f - zg) * ng + zg * hprev;
                float hv = (mask_s[row] > 0.0f) ? hn : hprev;
                hreg[jj][e] = hv;
                h_bf[row * 264 + col] = f2bf(hv);
            }
        }
        __syncthreads();

        if (seq_out) {
            uint16_t* dst = seq_out + ((size_t)b0 * NDIM + t) * 256;
            for (int i = tid; i < ROWS * 32; i += 512) {
                int r = i >> 5, c = i & 31;
                ((uint4*)(dst + (size_t)r * NDIM * 256))[c] =
                    ((const uint4*)(h_bf + r * 264))[c];
            }
        }
    }

    if (final_out) {  // final carry, full f32 precision from register state
#pragma unroll
        for (int jj = 0; jj < 2; ++jj) {
            int col = (ni * 2 + jj) * 16 + (lane & 15);
#pragma unroll
            for (int e = 0; e < 8; ++e) {
                int row = rowBase + e + ((lane >> 4) << 3);
                final_out[(size_t)(b0 + row) * 256 + col] = hreg[jj][e];
            }
        }
    }
}

// ---------------------------------------------------------------------------
extern "C" void kernel_launch(void* const* d_in, const int* in_sizes, int n_in,
                              void* d_out, int out_size, void* d_ws, size_t ws_size,
                              hipStream_t stream) {
    (void)in_sizes; (void)n_in; (void)out_size; (void)ws_size;

    const int*   pt_idx   = (const int*)d_in[0];
    const float* cont     = (const float*)d_in[1];
    const float* swing    = (const float*)d_in[2];
    const int*   sr_idx   = (const int*)d_in[3];
    const int*   bb_idx   = (const int*)d_in[4];
    const float* lspeed   = (const float*)d_in[5];
    const float* langle   = (const float*)d_in[6];
    const float* hc_x     = (const float*)d_in[7];
    const float* hc_y     = (const float*)d_in[8];
    const float* pmask    = (const float*)d_in[9];
    const float* amask    = (const float*)d_in[10];
    const float* pt_embed = (const float*)d_in[11];
    const float* sr_embed = (const float*)d_in[12];
    const float* bb_embed = (const float*)d_in[13];
    const float* W_ih_in  = (const float*)d_in[14];
    const float* W_hh_in  = (const float*)d_in[15];
    const float* b_ih_in  = (const float*)d_in[16];
    const float* b_hh_in  = (const float*)d_in[17];
    const float* W_ih_o0  = (const float*)d_in[18];
    const float* W_hh_o0  = (const float*)d_in[19];
    const float* b_ih_o0  = (const float*)d_in[20];
    const float* b_hh_o0  = (const float*)d_in[21];
    const float* W_ih_o1  = (const float*)d_in[22];
    const float* W_hh_o1  = (const float*)d_in[23];
    const float* b_ih_o1  = (const float*)d_in[24];
    const float* b_hh_o1  = (const float*)d_in[25];

    // ---- workspace layout (bf16 elements) ----
    uint16_t* ws16 = (uint16_t*)d_ws;
    const size_t FR = 512;  // bf16 per 32x16 fragment tile
    size_t o = 0;
    uint16_t* f_wih_in = ws16 + o; o += (size_t)48 * 1 * FR;
    uint16_t* f_whh_in = ws16 + o; o += (size_t)48 * 8 * FR;
    uint16_t* f_wih_o0 = ws16 + o; o += (size_t)48 * 9 * FR;
    uint16_t* f_whh_o0 = ws16 + o; o += (size_t)48 * 8 * FR;
    uint16_t* f_wih_o1 = ws16 + o; o += (size_t)48 * 8 * FR;
    uint16_t* f_whh_o1 = ws16 + o; o += (size_t)48 * 8 * FR;
    uint16_t* inner_out = ws16 + o; o += (size_t)BDIM * NDIM * 256;
    uint16_t* seq1      = ws16 + o;

    // ---- weight swizzle to WMMA B-fragment layout (bf16) ----
    auto swz = [&](const float* W, int Kreal, int KT, uint16_t* dst) {
        int total = 48 * KT * 512;
        swizzle_weight_kernel<<<(total + 255) / 256, 256, 0, stream>>>(W, Kreal, KT, dst, total);
    };
    swz(W_ih_in, 29, 1, f_wih_in);
    swz(W_hh_in, 256, 8, f_whh_in);
    swz(W_ih_o0, 264, 9, f_wih_o0);
    swz(W_hh_o0, 256, 8, f_whh_o0);
    swz(W_ih_o1, 256, 8, f_wih_o1);
    swz(W_hh_o1, 256, 8, f_whh_o1);

    // ---- inner GRU over pitches ----
    inner_gru_kernel<<<(BDIM * NDIM) / ROWS, 512, 0, stream>>>(
        pt_idx, cont, swing, sr_idx, pmask, pt_embed, sr_embed,
        f_wih_in, f_whh_in, b_ih_in, b_hh_in, inner_out);

    // ---- outer GRU layer 0 (with extras, writes full sequence) ----
    outer_gru_kernel<<<BDIM / ROWS, 512, 0, stream>>>(
        inner_out, bb_idx, bb_embed, lspeed, langle, hc_x, hc_y, /*has_extra=*/1,
        amask, f_wih_o0, /*ktx=*/9, f_whh_o0, b_ih_o0, b_hh_o0,
        seq1, /*final_out=*/nullptr);

    // ---- outer GRU layer 1 (final hidden state -> d_out f32 [256,256]) ----
    outer_gru_kernel<<<BDIM / ROWS, 512, 0, stream>>>(
        seq1, nullptr, nullptr, nullptr, nullptr, nullptr, nullptr, /*has_extra=*/0,
        amask, f_wih_o1, /*ktx=*/8, f_whh_o1, b_ih_o1, b_hh_o1,
        /*seq_out=*/nullptr, (float*)d_out);
}
